// CrossAttention_79362405696071
// MI455X (gfx1250) — matrix operand order repaired and checked by reading, not compiled, not run on hardware
//
#include <hip/hip_runtime.h>
#include <hip/hip_bf16.h>

// ---------------------------------------------------------------------------
// Problem constants (from reference)
// ---------------------------------------------------------------------------
#define DIMX   512
#define INNER  512
#define HEADS  8
#define DHEAD  64
#define BATCH  2
#define NSEQ   2048
#define SCALE  0.125f   // DIM_HEAD ** -0.5 = 64^-0.5

typedef unsigned short u16;
typedef __attribute__((ext_vector_type(16))) __bf16        v16bf;
typedef __attribute__((ext_vector_type(8)))  float         v8f;
typedef __attribute__((ext_vector_type(4)))  unsigned int  u32x4;

union FragU { v16bf v; u32x4 q[2]; };
union V16U  { u16 h[16]; u32x4 q[2]; };

// f32 -> bf16, round-to-nearest-even
__device__ __forceinline__ u16 to_bf16(float f) {
    unsigned u = __float_as_uint(f);
    unsigned r = u + 0x7FFFu + ((u >> 16) & 1u);
    return (u16)(r >> 16);
}

// ---------------------------------------------------------------------------
// WMMA fragment loaders (CDNA5 16-bit layouts, wave32)
//   A (16x32, MxK): lane holds row m=lane&15; hi=lane>>4 selects K interleave:
//     VGPR0-3 -> K = hi*8 .. hi*8+7 ; VGPR4-7 -> K = 16+hi*8 .. +7
//   B (32x16, KxN): lane holds col n=lane&15; same K interleave as A
// ---------------------------------------------------------------------------
__device__ __forceinline__ v16bf load_frag_a(const u16* base, int ld) {
    int lane = threadIdx.x & 31;
    int row  = lane & 15;
    int hi   = lane >> 4;
    FragU f;
    const u16* p = base + (size_t)row * ld;
    f.q[0] = *(const u32x4*)(p + hi * 8);
    f.q[1] = *(const u32x4*)(p + 16 + hi * 8);
    return f.v;
}

__device__ __forceinline__ v16bf load_frag_b(const u16* base, int ld) {
    int lane = threadIdx.x & 31;
    int col  = lane & 15;
    int hi   = lane >> 4;
    FragU f;
    const u16* p = base + (size_t)col * ld + hi * 8;
    f.q[0] = *(const u32x4*)(p);
    f.q[1] = *(const u32x4*)(p + 16);
    return f.v;
}

__device__ __forceinline__ v8f wmma_bf16(v16bf a, v16bf b, v8f c) {
    return __builtin_amdgcn_wmma_f32_16x16x32_bf16(false, a, false, b,
                                                   (short)0, c, false, false);
}

// ---------------------------------------------------------------------------
// Elementwise conversion / weight transpose
// ---------------------------------------------------------------------------
__global__ void cvt_bf16_kernel(const float* __restrict__ in,
                                u16* __restrict__ out, int n) {
    int i = blockIdx.x * blockDim.x + threadIdx.x;
    if (i < n) out[i] = to_bf16(in[i]);
}

// W[in=512][out=512] f32  ->  Wt[out=512][in=512] bf16
__global__ void wtrans_kernel(const float* __restrict__ W, u16* __restrict__ Wt) {
    int i = blockIdx.x * blockDim.x + threadIdx.x;  // 0 .. 512*512-1
    int o = i >> 9;
    int k = i & 511;
    Wt[(size_t)o * 512 + k] = to_bf16(W[(size_t)k * 512 + o]);
}

// ---------------------------------------------------------------------------
// bf16 WMMA GEMM: C[M,512] = A[M,512] @ W[512,512]   (W given transposed: Bt[n][k])
//   grid.x = M/128, grid.y = 512/128; block = 256 (8 waves)
//   wave (wy 0..3, wx 0..1): 32x64 tile, software-pipelined fragment loads.
//   Cf != null -> f32 output + bias; else bf16 output to Cb.
// ---------------------------------------------------------------------------
__global__ void __launch_bounds__(256) gemm_bf16_kernel(
        const u16* __restrict__ A, const u16* __restrict__ Bt,
        u16* __restrict__ Cb, float* __restrict__ Cf,
        const float* __restrict__ bias) {
    int wave = threadIdx.x >> 5;
    int lane = threadIdx.x & 31;
    int wy = wave & 3, wx = wave >> 2;
    int m0 = blockIdx.x * 128 + wy * 32;
    int n0 = blockIdx.y * 128 + wx * 64;

    const u16* A0 = A + (size_t)m0 * 512;
    const u16* A1 = A + (size_t)(m0 + 16) * 512;
    const u16* B0 = Bt + (size_t)(n0 +  0) * 512;
    const u16* B1 = Bt + (size_t)(n0 + 16) * 512;
    const u16* B2 = Bt + (size_t)(n0 + 32) * 512;
    const u16* B3 = Bt + (size_t)(n0 + 48) * 512;

    v8f acc0[4] = {}, acc1[4] = {};

    v16bf a0 = load_frag_a(A0, 512);
    v16bf a1 = load_frag_a(A1, 512);
    v16bf b0 = load_frag_b(B0, 512);
    v16bf b1 = load_frag_b(B1, 512);
    v16bf b2 = load_frag_b(B2, 512);
    v16bf b3 = load_frag_b(B3, 512);

    for (int kk = 0; kk < 512; kk += 32) {
        int kn = kk + 32;
        v16bf a0n = a0, a1n = a1, b0n = b0, b1n = b1, b2n = b2, b3n = b3;
        if (kn < 512) {  // issue next-step loads before this step's WMMAs
            a0n = load_frag_a(A0 + kn, 512);
            a1n = load_frag_a(A1 + kn, 512);
            b0n = load_frag_b(B0 + kn, 512);
            b1n = load_frag_b(B1 + kn, 512);
            b2n = load_frag_b(B2 + kn, 512);
            b3n = load_frag_b(B3 + kn, 512);
        }
        acc0[0] = wmma_bf16(a0, b0, acc0[0]);
        acc0[1] = wmma_bf16(a0, b1, acc0[1]);
        acc0[2] = wmma_bf16(a0, b2, acc0[2]);
        acc0[3] = wmma_bf16(a0, b3, acc0[3]);
        acc1[0] = wmma_bf16(a1, b0, acc1[0]);
        acc1[1] = wmma_bf16(a1, b1, acc1[1]);
        acc1[2] = wmma_bf16(a1, b2, acc1[2]);
        acc1[3] = wmma_bf16(a1, b3, acc1[3]);
        a0 = a0n; a1 = a1n; b0 = b0n; b1 = b1n; b2 = b2n; b3 = b3n;
    }

    int col = lane & 15, hi = lane >> 4;
#pragma unroll
    for (int half = 0; half < 2; ++half) {
        v8f* accp = half ? acc1 : acc0;
        int mb = m0 + half * 16;
        if (Cf) {
#pragma unroll
            for (int j = 0; j < 4; ++j) {
                int n = n0 + j * 16 + col;
                float bv = bias ? bias[n] : 0.0f;
#pragma unroll
                for (int r = 0; r < 8; ++r)
                    Cf[(size_t)(mb + r + 8 * hi) * 512 + n] = accp[j][r] + bv;
            }
        } else {
#pragma unroll
            for (int j = 0; j < 4; ++j) {
                int n = n0 + j * 16 + col;
#pragma unroll
                for (int r = 0; r < 8; ++r)
                    Cb[(size_t)(mb + r + 8 * hi) * 512 + n] = to_bf16(accp[j][r]);
            }
        }
    }
}

// ---------------------------------------------------------------------------
// Fused dual-pass flash attention, 64-key blocks, double-buffered LDS.
//   out(q-mod) = 0.5 * ( attend(q, K_cross, V_cross) + attend(q, K_self, V_self) )
//   appended zero key/value slot handled analytically per pass.
//   grid = (NSEQ/128, HEADS, 2*BATCH); block = 256 (8 waves, 16 query rows each)
// ---------------------------------------------------------------------------
__global__ void __launch_bounds__(256) attn_kernel(
        const u16* __restrict__ Qi, const u16* __restrict__ Ki, const u16* __restrict__ Vi,
        const u16* __restrict__ Qc, const u16* __restrict__ Kc, const u16* __restrict__ Vc,
        u16* __restrict__ Oi, u16* __restrict__ Oc) {
    __shared__ __attribute__((aligned(16))) u16 Klds [2][64 * 64];  // K rows
    __shared__ __attribute__((aligned(16))) u16 Vtlds[2][64 * 64];  // V transposed
    __shared__ __attribute__((aligned(16))) u16 Plds [8][16 * 72];  // per-wave P

    int tid  = threadIdx.x;
    int wave = tid >> 5, lane = tid & 31;
    int col  = lane & 15, hi = lane >> 4;

    int qt  = blockIdx.x;
    int h   = blockIdx.y;
    int b   = blockIdx.z & 1;
    int mod = blockIdx.z >> 1;

    const u16* Q      = mod ? Qc : Qi;
    const u16* Kself  = mod ? Kc : Ki;
    const u16* Vself  = mod ? Vc : Vi;
    const u16* Kcross = mod ? Ki : Kc;
    const u16* Vcross = mod ? Vi : Vc;
    u16*       Out    = mod ? Oc : Oi;

    // stage one 64-key block (K rows + V transposed) into buffer `buf`
    auto stage = [&](int buf, int kb, const u16* Kg, const u16* Vg) {
        // K tile: 64 keys x 64 d, straight b128 copies
        int row = tid >> 2;
        int seg = (tid & 3) * 16;
        const u16* kp = Kg + ((size_t)b * NSEQ + kb + row) * INNER + h * DHEAD + seg;
        u16* kd = Klds[buf] + row * 64 + seg;
        *(u32x4*)(kd)     = *(const u32x4*)(kp);
        *(u32x4*)(kd + 8) = *(const u32x4*)(kp + 8);
        if (kb + 128 < NSEQ)  // pull the block after next toward L2/L0
            __builtin_prefetch(kp + (size_t)128 * INNER, 0, 0);
        // V transposed: lane-coalesced global u16 reads, b128 LDS writes
        int d  = tid & 63;
        int kg = (tid >> 6) * 16;
        const u16* vp = Vg + ((size_t)b * NSEQ + kb + kg) * INNER + h * DHEAD + d;
        V16U tv;
#pragma unroll
        for (int i = 0; i < 16; ++i) tv.h[i] = vp[(size_t)i * INNER];
        u16* vd = Vtlds[buf] + d * 64 + kg;
        *(u32x4*)(vd)     = tv.q[0];
        *(u32x4*)(vd + 8) = tv.q[1];
    };

    // Q fragments (16 rows x 64 d), held in VGPRs for both passes
    size_t baseQ = ((size_t)b * NSEQ + (size_t)qt * 128 + wave * 16) * INNER + h * DHEAD;
    v16bf qf0 = load_frag_a(Q + baseQ, INNER);
    v16bf qf1 = load_frag_a(Q + baseQ + 32, INNER);

    v8f out_acc[4] = {};
    u16* Pw = Plds[wave];

    for (int pass = 0; pass < 2; ++pass) {
        const u16* Kg = (pass == 0) ? Kcross : Kself;
        const u16* Vg = (pass == 0) ? Vcross : Vself;

        float m_run[8], l_run[8];
#pragma unroll
        for (int r = 0; r < 8; ++r) { m_run[r] = -1e30f; l_run[r] = 0.0f; }
        v8f acc[4] = {};

        stage(0, 0, Kg, Vg);
        __syncthreads();

        int ib = 0;
        for (int kb = 0; kb < NSEQ; kb += 64, ib ^= 1) {
            if (kb + 64 < NSEQ) stage(ib ^ 1, kb + 64, Kg, Vg);  // overlap next stage

            const u16* Kl  = Klds[ib];
            const u16* Vtl = Vtlds[ib];

            // ---- S = Q K^T : four 16x16 blocks over 64 keys (8 WMMAs)
            v8f s[4] = {};
#pragma unroll
            for (int c = 0; c < 2; ++c) {
                v16bf qf = c ? qf1 : qf0;
#pragma unroll
                for (int sub = 0; sub < 4; ++sub) {
                    v16bf kf = load_frag_b(Kl + sub * 16 * 64 + c * 32, 64);
                    s[sub] = wmma_bf16(qf, kf, s[sub]);
                }
            }

            // ---- online softmax over the 64 new keys
            float t[4][8], bm[8];
#pragma unroll
            for (int r = 0; r < 8; ++r) {
#pragma unroll
                for (int sub = 0; sub < 4; ++sub) t[sub][r] = s[sub][r] * SCALE;
                bm[r] = fmaxf(fmaxf(t[0][r], t[1][r]), fmaxf(t[2][r], t[3][r]));
            }
#pragma unroll
            for (int off = 1; off < 16; off <<= 1)
#pragma unroll
                for (int r = 0; r < 8; ++r)
                    bm[r] = fmaxf(bm[r], __shfl_xor(bm[r], off, 32));

            float rs[8];
#pragma unroll
            for (int r = 0; r < 8; ++r) {
                float mn = fmaxf(m_run[r], bm[r]);
                float al = __expf(m_run[r] - mn);
                float p0 = __expf(t[0][r] - mn);
                float p1 = __expf(t[1][r] - mn);
                float p2 = __expf(t[2][r] - mn);
                float p3 = __expf(t[3][r] - mn);
                m_run[r] = mn;
                rs[r]    = (p0 + p1) + (p2 + p3);
                l_run[r] *= al;
                int prow = (r + 8 * hi) * 72;
                Pw[prow + col]      = to_bf16(p0);
                Pw[prow + 16 + col] = to_bf16(p1);
                Pw[prow + 32 + col] = to_bf16(p2);
                Pw[prow + 48 + col] = to_bf16(p3);
#pragma unroll
                for (int j = 0; j < 4; ++j) acc[j][r] *= al;
            }
#pragma unroll
            for (int off = 1; off < 16; off <<= 1)
#pragma unroll
                for (int r = 0; r < 8; ++r) rs[r] += __shfl_xor(rs[r], off, 32);
#pragma unroll
            for (int r = 0; r < 8; ++r) l_run[r] += rs[r];

            // ---- O += P V  (8 WMMAs; same-wave DS ordering, compile barrier only)
            asm volatile("" ::: "memory");
            v16bf pf0 = load_frag_a(Pw, 72);
            v16bf pf1 = load_frag_a(Pw + 32, 72);
#pragma unroll
            for (int j = 0; j < 4; ++j) {
                v16bf v0 = load_frag_b(Vtl + (j * 16) * 64, 64);
                v16bf v1 = load_frag_b(Vtl + (j * 16) * 64 + 32, 64);
                acc[j] = wmma_bf16(pf0, v0, acc[j]);
                acc[j] = wmma_bf16(pf1, v1, acc[j]);
            }

            __syncthreads();  // all waves done with buffer ib before restaging it
        }

        // ---- appended zero key/value slot: logit 0, value 0
#pragma unroll
        for (int r = 0; r < 8; ++r) {
            float mn = fmaxf(m_run[r], 0.0f);
            float al = __expf(m_run[r] - mn);
            float l  = l_run[r] * al + __expf(-mn);
            float sc = al / l;
#pragma unroll
            for (int j = 0; j < 4; ++j) out_acc[j][r] += acc[j][r] * sc;
        }
    }

    // ---- store 0.5*(cross+self) in [B, N, H*DHEAD] bf16
    size_t baseO = (size_t)b * NSEQ + (size_t)qt * 128 + wave * 16;
#pragma unroll
    for (int j = 0; j < 4; ++j)
#pragma unroll
        for (int r = 0; r < 8; ++r) {
            size_t n = baseO + r + 8 * hi;
            Out[n * INNER + h * DHEAD + j * 16 + col] = to_bf16(out_acc[j][r] * 0.5f);
        }
}

// ---------------------------------------------------------------------------
// Host launcher
// ---------------------------------------------------------------------------
extern "C" void kernel_launch(void* const* d_in, const int* in_sizes, int n_in,
                              void* d_out, int out_size, void* d_ws, size_t ws_size,
                              hipStream_t stream) {
    const float* img  = (const float*)d_in[0];
    const float* clin = (const float*)d_in[1];
    const float* Wsrc[8] = {
        (const float*)d_in[2],  // W_iq
        (const float*)d_in[3],  // W_ik
        (const float*)d_in[4],  // W_iv
        (const float*)d_in[5],  // W_cq
        (const float*)d_in[6],  // W_ck
        (const float*)d_in[7],  // W_cv
        (const float*)d_in[8],  // W_io
        (const float*)d_in[10], // W_co
    };
    const float* b_io = (const float*)d_in[9];
    const float* b_co = (const float*)d_in[11];

    const size_t NTOK  = (size_t)BATCH * NSEQ;  // 4096
    const size_t TELEM = NTOK * DIMX;           // 2,097,152 elements
    const size_t WELEM = 512 * 512;             // 262,144 elements

    u16* ws = (u16*)d_ws;
    u16* img_b  = ws; ws += TELEM;
    u16* clin_b = ws; ws += TELEM;
    u16* Wt[8];
    for (int i = 0; i < 8; ++i) { Wt[i] = ws; ws += WELEM; }
    u16* Qi = ws; ws += TELEM;
    u16* Ki = ws; ws += TELEM;
    u16* Vi = ws; ws += TELEM;
    u16* Qc = ws; ws += TELEM;
    u16* Kc = ws; ws += TELEM;
    u16* Vc = ws; ws += TELEM;
    u16* Ai = ws; ws += TELEM;
    u16* Ac = ws; ws += TELEM;

    // Stage 0: bf16 conversion + weight transpose
    int nT = (int)TELEM;
    cvt_bf16_kernel<<<(nT + 255) / 256, 256, 0, stream>>>(img, img_b, nT);
    cvt_bf16_kernel<<<(nT + 255) / 256, 256, 0, stream>>>(clin, clin_b, nT);
    for (int i = 0; i < 8; ++i)
        wtrans_kernel<<<(int)(WELEM / 256), 256, 0, stream>>>(Wsrc[i], Wt[i]);

    // Stage 1: QKV projections (M=4096, K=N=512)
    dim3 gg(4096 / 128, 512 / 128);
    gemm_bf16_kernel<<<gg, 256, 0, stream>>>(img_b,  Wt[0], Qi, nullptr, nullptr);
    gemm_bf16_kernel<<<gg, 256, 0, stream>>>(img_b,  Wt[1], Ki, nullptr, nullptr);
    gemm_bf16_kernel<<<gg, 256, 0, stream>>>(img_b,  Wt[2], Vi, nullptr, nullptr);
    gemm_bf16_kernel<<<gg, 256, 0, stream>>>(clin_b, Wt[3], Qc, nullptr, nullptr);
    gemm_bf16_kernel<<<gg, 256, 0, stream>>>(clin_b, Wt[4], Kc, nullptr, nullptr);
    gemm_bf16_kernel<<<gg, 256, 0, stream>>>(clin_b, Wt[5], Vc, nullptr, nullptr);

    // Stage 2: fused cross+self flash attention (both modalities)
    attn_kernel<<<dim3(NSEQ / 128, HEADS, 2 * BATCH), 256, 0, stream>>>(
        Qi, Ki, Vi, Qc, Kc, Vc, Ai, Ac);

    // Stage 3: output projections (f32 + bias) directly into d_out
    float* out_f = (float*)d_out;
    gemm_bf16_kernel<<<gg, 256, 0, stream>>>(Ai, Wt[6], nullptr, out_f,         b_io);
    gemm_bf16_kernel<<<gg, 256, 0, stream>>>(Ac, Wt[7], nullptr, out_f + TELEM, b_co);
}
